// MultiHeadAttention_1735166788682
// MI455X (gfx1250) — matrix operand order repaired and checked by reading, not compiled, not run on hardware
//
#include <hip/hip_runtime.h>
#include <cstdint>
#include <cstddef>

// ---------------------------------------------------------------------------
// MHA forward for MI455X (gfx1250, wave32, WMMA).
// All GEMMs use v_wmma_f32_16x16x32_f16 (f16 operands, fp32 accum).
// ---------------------------------------------------------------------------

#define NSEQ   4096
#define DMODEL 400
#define DPAD   416          // 400 padded to 13*32 for K loops
#define NHEAD  4
#define NPR    16384        // quirky rows: 4096*4
#define HKP    128          // per-head dim 100 padded to 4*32
#define HD     100

typedef _Float16 half8 __attribute__((ext_vector_type(8)));
typedef _Float16 v16h  __attribute__((ext_vector_type(16)));
typedef float    v8f   __attribute__((ext_vector_type(8)));
typedef float    f32x8 __attribute__((ext_vector_type(8)));

static __device__ inline v8f zero8() {
  v8f z;
#pragma unroll
  for (int i = 0; i < 8; ++i) z[i] = 0.0f;
  return z;
}

static __device__ inline v16h combine16(half8 lo, half8 hi) {
  v16h r;
#pragma unroll
  for (int i = 0; i < 8; ++i) { r[i] = lo[i]; r[i + 8] = hi[i]; }
  return r;
}

// A-fragment (16xK tile @ row0,k0) of an f16 row-major matrix, stride in halves.
// Lane L (m=L&15, h=L>>4): elems 0..7 -> K = k0+8h+e ; elems 8..15 -> K = k0+16+8h+(e-8)
static __device__ inline v16h load_a_f16(const _Float16* __restrict__ base, int stride,
                                         int row0, int k0, int lane) {
  int m = lane & 15, hf = lane >> 4;
  const _Float16* p = base + (size_t)(row0 + m) * stride;
  half8 lo = *(const half8*)(p + k0 + hf * 8);
  half8 hi = *(const half8*)(p + k0 + 16 + hf * 8);
  return combine16(lo, hi);
}

// Same A-fragment but from an fp32 source, converted to f16 in registers.
static __device__ inline v16h load_a_f32(const float* __restrict__ base, int stride,
                                         int row0, int k0, int lane) {
  int m = lane & 15, hf = lane >> 4;
  const float* p = base + (size_t)(row0 + m) * stride;
  f32x8 lo = *(const f32x8*)(p + k0 + hf * 8);
  f32x8 hi = *(const f32x8*)(p + k0 + 16 + hf * 8);
  v16h r;
#pragma unroll
  for (int i = 0; i < 8; ++i) { r[i] = (_Float16)lo[i]; r[i + 8] = (_Float16)hi[i]; }
  return r;
}

// B-fragment (Kx16): B[kk][nn] = src[(n0+nn)*stride + k0+kk]   (B = rows^T)
// Lane L (n=L&15, h=L>>4): elems 0..7 -> K = k0+8h+e ; elems 8..15 -> K = k0+16+8h+(e-8)
static __device__ inline v16h load_b_rows(const _Float16* __restrict__ base, int stride,
                                          int n0, int k0, int lane) {
  int n = lane & 15, hf = lane >> 4;
  const _Float16* p = base + (size_t)(n0 + n) * stride + k0 + hf * 8;
  half8 lo = *(const half8*)(p);
  half8 hi = *(const half8*)(p + 16);
  return combine16(lo, hi);
}

static __device__ inline v8f wmma16(v16h a, v16h b, v8f c) {
  return __builtin_amdgcn_wmma_f32_16x16x32_f16(false, a, false, b, (short)0, c, false, false);
}

// ---------------------------------------------------------------------------
// Kernel 0a: fp32 -> f16 with column zero-padding (for X and W matrices).
// ---------------------------------------------------------------------------
__global__ void cvt_pad_kernel(const float* __restrict__ src, _Float16* __restrict__ dst,
                               int rows, int cs, int cd) {
  int n = rows * cd;
  for (int i = blockIdx.x * blockDim.x + threadIdx.x; i < n; i += gridDim.x * blockDim.x) {
    int r = i / cd, c = i - r * cd;
    dst[i] = (_Float16)((c < cs) ? src[(size_t)r * cs + c] : 0.0f);
  }
}

// Kernel 0b: zero a byte region (16B granules). Clears Qp/Kp/Vt/CtxG incl. pads.
__global__ void zero16_kernel(uint4* __restrict__ p, int n) {
  uint4 z; z.x = 0u; z.y = 0u; z.z = 0u; z.w = 0u;
  for (int i = blockIdx.x * blockDim.x + threadIdx.x; i < n; i += gridDim.x * blockDim.x)
    p[i] = z;
}

// ---------------------------------------------------------------------------
// Kernel 1: projection  P = X @ W^T + b  (f16 WMMA, fp32 accum).
// mode Q/K: store into quirky [16384,128] row-major (d-padded).
// mode V  : store transposed Vt[128][16384].
// Block: 128 threads (4 waves), one 16-row M-tile per block, waves split 25 N-tiles.
// ---------------------------------------------------------------------------
__global__ __launch_bounds__(128) void proj_kernel(
    const _Float16* __restrict__ X,    // [4096,416] f16
    const _Float16* __restrict__ W,    // [400,416]  f16
    const float* __restrict__ bias,    // [400]
    _Float16* __restrict__ dstQK,      // [16384,128] or nullptr
    _Float16* __restrict__ dstVt) {    // [128,16384] or nullptr
  int wv = threadIdx.x >> 5, lane = threadIdx.x & 31;
  int m0 = blockIdx.x * 16;

  v8f acc[7];
#pragma unroll
  for (int i = 0; i < 7; ++i) acc[i] = zero8();

  for (int ks = 0; ks < 13; ++ks) {
    int k0 = ks * 32;
    v16h a = load_a_f16(X, DPAD, m0, k0, lane);
#pragma unroll
    for (int i = 0; i < 7; ++i) {
      int t = wv + 4 * i; t = (t < 25) ? t : 24;   // clamp: redundant but uniform
      v16h b = load_b_rows(W, DPAD, t * 16, k0, lane);
      acc[i] = wmma16(a, b, acc[i]);
    }
  }

  int hf = lane >> 4, nn = lane & 15;
#pragma unroll
  for (int i = 0; i < 7; ++i) {
    int t = wv + 4 * i;
    if (t < 25) {
      int o = t * 16 + nn;
      float bo = bias[o];
#pragma unroll
      for (int r = 0; r < 8; ++r) {
        int row = m0 + hf * 8 + r;                  // row in [4096,400] view
        float val = acc[i][r] + bo;
        int np = 4 * row + o / 100;                 // quirky row in [16384,100]
        int j  = o % 100;
        if (dstQK) dstQK[(size_t)np * HKP + j] = (_Float16)val;
        else       dstVt[(size_t)j * NPR + np] = (_Float16)val;
      }
    }
  }
}

// ---------------------------------------------------------------------------
// Kernel 2: scores (q@k^T * 0.2) + softmax, fused in LDS, write attn fp32.
// Grid (256 strips, 4 heads), 256 threads (8 waves). 263 KB dynamic LDS:
// full 16x4096 fp32 score strip resident (CDNA5: 320 KB/WGP).
// ---------------------------------------------------------------------------
__global__ __launch_bounds__(256) void attn_kernel(
    const _Float16* __restrict__ Qp,   // [16384,128]
    const _Float16* __restrict__ Kp,   // [16384,128]
    float* __restrict__ attn) {        // [4,4096,4096]
  extern __shared__ float smem[];
  float* S      = smem;                      // 16*4096
  float* red    = smem + 16 * 4096;          // 16*16
  float* rowmax = red + 256;                 // 16
  float* rowinv = rowmax + 16;               // 16

  int wv = threadIdx.x >> 5, lane = threadIdx.x & 31;
  int h = blockIdx.y;
  int row0 = blockIdx.x * 16;
  const _Float16* Qb = Qp + (size_t)h * 4096 * HKP;
  const _Float16* Kb = Kp + (size_t)h * 4096 * HKP;

  // A fragments for this 16-row strip (K = 128 padded head dim, 4 steps)
  v16h aq[4];
#pragma unroll
  for (int ks = 0; ks < 4; ++ks) aq[ks] = load_a_f16(Qb, HKP, row0, ks * 32, lane);

  int hf = lane >> 4, nn = lane & 15;
  for (int ct = wv; ct < 256; ct += 8) {
    int col0 = ct * 16;
    v8f acc = zero8();
#pragma unroll
    for (int ks = 0; ks < 4; ++ks) {
      v16h b = load_b_rows(Kb, HKP, col0, ks * 32, lane);
      acc = wmma16(aq[ks], b, acc);
    }
#pragma unroll
    for (int r = 0; r < 8; ++r)
      S[(hf * 8 + r) * 4096 + col0 + nn] = acc[r] * 0.2f;   // quirky scale 25^-0.5
  }
  __syncthreads();

  // softmax over 4096 cols; 16 threads per row
  int r = threadIdx.x >> 4, t = threadIdx.x & 15;
  float m = -3.402823e38f;
  for (int c = t; c < 4096; c += 16) m = fmaxf(m, S[r * 4096 + c]);
  red[r * 16 + t] = m;
  __syncthreads();
  if (t == 0) {
    float mm = red[r * 16];
#pragma unroll
    for (int i = 1; i < 16; ++i) mm = fmaxf(mm, red[r * 16 + i]);
    rowmax[r] = mm;
  }
  __syncthreads();
  float rm = rowmax[r];
  float s = 0.0f;
  for (int c = t; c < 4096; c += 16) {
    float e = __expf(S[r * 4096 + c] - rm);
    S[r * 4096 + c] = e;
    s += e;
  }
  red[r * 16 + t] = s;
  __syncthreads();
  if (t == 0) {
    float ss = 0.0f;
#pragma unroll
    for (int i = 0; i < 16; ++i) ss += red[r * 16 + i];
    rowinv[r] = 1.0f / ss;
  }
  __syncthreads();

  // coalesced streaming write of normalized attn
  float* out = attn + ((size_t)h << 24) + (size_t)row0 * 4096;
  for (int idx = threadIdx.x; idx < 16 * 4096; idx += 256) {
    int rr = idx >> 12, c = idx & 4095;
    out[(size_t)rr * 4096 + c] = S[rr * 4096 + c] * rowinv[rr];
  }
}

// ---------------------------------------------------------------------------
// Kernel 3: ctx = attn @ v. A from fp32 attn (converted in regs), B from Vt.
// Stores ctx f16 directly in the reshaped-back [4096,416] layout (CtxG).
// Grid (64,4), 128 threads = 4 waves, one 16-row M-tile per wave, 7 N-tiles.
// ---------------------------------------------------------------------------
__global__ __launch_bounds__(128) void ctx_kernel(
    const float* __restrict__ attn,    // [4,4096,4096]
    const _Float16* __restrict__ Vt,   // [128,16384] (rows 100..127 zero)
    _Float16* __restrict__ CtxG) {     // [4096,416]
  int wv = threadIdx.x >> 5, lane = threadIdx.x & 31;
  int h = blockIdx.y;
  int m0 = (blockIdx.x * 4 + wv) * 16;
  const float* Ab = attn + ((size_t)h << 24);
  const _Float16* Vb = Vt + (size_t)h * 4096;

  v8f acc[7];
#pragma unroll
  for (int i = 0; i < 7; ++i) acc[i] = zero8();

  for (int ks = 0; ks < 128; ++ks) {
    int k0 = ks * 32;
    v16h a = load_a_f32(Ab, 4096, m0, k0, lane);
#pragma unroll
    for (int t = 0; t < 7; ++t) {
      v16h b = load_b_rows(Vb, NPR, t * 16, k0, lane);
      acc[t] = wmma16(a, b, acc[t]);
    }
  }

  int hf = lane >> 4, nn = lane & 15;
#pragma unroll
  for (int t = 0; t < 7; ++t) {
    int j = t * 16 + nn;
    if (j < 100) {
#pragma unroll
      for (int r = 0; r < 8; ++r) {
        int pos = m0 + hf * 8 + r;                         // quirky seq pos in head
        size_t row = (size_t)(h * 1024 + (pos >> 2));      // row of [4096,400] view
        int col = (pos & 3) * 100 + j;
        CtxG[row * DPAD + col] = (_Float16)acc[t][r];
      }
    }
  }
}

// ---------------------------------------------------------------------------
// Kernel 4: out = ctx @ Wo^T + bo ; x = residual + out ; LayerNorm(x).
// Grid 256, 128 threads (4 waves), 16 rows per block, row stats in LDS.
// ---------------------------------------------------------------------------
__global__ __launch_bounds__(128) void outln_kernel(
    const _Float16* __restrict__ CtxG,   // [4096,416]
    const _Float16* __restrict__ Wo16,   // [400,416]
    const float* __restrict__ bo,        // [400]
    const float* __restrict__ residual,  // query fp32 [4096,400]
    const float* __restrict__ gamma, const float* __restrict__ beta,
    float* __restrict__ out) {           // [4096,400]
  __shared__ float xbuf[16 * DMODEL];
  __shared__ float red[16 * 8];
  __shared__ float mu[16], rs[16];

  int wv = threadIdx.x >> 5, lane = threadIdx.x & 31;
  int m0 = blockIdx.x * 16;

  v8f acc[7];
#pragma unroll
  for (int i = 0; i < 7; ++i) acc[i] = zero8();

  for (int ks = 0; ks < 13; ++ks) {
    int k0 = ks * 32;
    v16h a = load_a_f16(CtxG, DPAD, m0, k0, lane);
#pragma unroll
    for (int i = 0; i < 7; ++i) {
      int t = wv + 4 * i; t = (t < 25) ? t : 24;
      v16h b = load_b_rows(Wo16, DPAD, t * 16, k0, lane);
      acc[i] = wmma16(a, b, acc[i]);
    }
  }

  int hf = lane >> 4, nn = lane & 15;
#pragma unroll
  for (int i = 0; i < 7; ++i) {
    int t = wv + 4 * i;
    if (t < 25) {
      int o = t * 16 + nn;
      float bb = bo[o];
#pragma unroll
      for (int r = 0; r < 8; ++r) {
        int lr = hf * 8 + r;
        xbuf[lr * DMODEL + o] = acc[i][r] + bb + residual[(size_t)(m0 + lr) * DMODEL + o];
      }
    }
  }
  __syncthreads();

  // LayerNorm: 8 threads per row
  int r = threadIdx.x >> 3, t = threadIdx.x & 7;
  float s = 0.0f;
  for (int c = t; c < DMODEL; c += 8) s += xbuf[r * DMODEL + c];
  red[r * 8 + t] = s;
  __syncthreads();
  if (t == 0) {
    float ss = 0.0f;
#pragma unroll
    for (int i = 0; i < 8; ++i) ss += red[r * 8 + i];
    mu[r] = ss * (1.0f / DMODEL);
  }
  __syncthreads();
  float mean = mu[r];
  s = 0.0f;
  for (int c = t; c < DMODEL; c += 8) {
    float d = xbuf[r * DMODEL + c] - mean;
    s += d * d;
  }
  red[r * 8 + t] = s;
  __syncthreads();
  if (t == 0) {
    float ss = 0.0f;
#pragma unroll
    for (int i = 0; i < 8; ++i) ss += red[r * 8 + i];
    rs[r] = rsqrtf(ss * (1.0f / DMODEL) + 1e-5f);
  }
  __syncthreads();

  for (int idx = threadIdx.x; idx < 16 * DMODEL; idx += 128) {
    int lr = idx / DMODEL, o = idx - lr * DMODEL;
    float y = (xbuf[idx] - mu[lr]) * rs[lr] * gamma[o] + beta[o];
    out[(size_t)(m0 + lr) * DMODEL + o] = y;
  }
}

// ---------------------------------------------------------------------------
// Host launcher
// ---------------------------------------------------------------------------
extern "C" void kernel_launch(void* const* d_in, const int* in_sizes, int n_in,
                              void* d_out, int out_size, void* d_ws, size_t ws_size,
                              hipStream_t stream) {
  (void)in_sizes; (void)n_in; (void)out_size; (void)ws_size;
  const float* key   = (const float*)d_in[0];
  const float* value = (const float*)d_in[1];
  const float* query = (const float*)d_in[2];
  const float* Wq = (const float*)d_in[3];  const float* bq = (const float*)d_in[4];
  const float* Wk = (const float*)d_in[5];  const float* bk = (const float*)d_in[6];
  const float* Wv = (const float*)d_in[7];  const float* bv = (const float*)d_in[8];
  const float* Wo = (const float*)d_in[9];  const float* bo = (const float*)d_in[10];
  const float* gamma = (const float*)d_in[11];
  const float* beta  = (const float*)d_in[12];

  float* out  = (float*)d_out;
  float* attn = out + (size_t)NSEQ * DMODEL;   // [4,4096,4096] fp32

  // workspace layout (all sizes are 256B multiples -> contiguous)
  char* ws = (char*)d_ws;
  size_t off = 0;
  auto carve = [&](size_t bytes) { void* p = ws + off; off += (bytes + 255) & ~(size_t)255; return p; };
  const size_t XB = (size_t)NSEQ * DPAD * 2;     // 3,407,872
  const size_t WB = (size_t)DMODEL * DPAD * 2;   //   332,800
  const size_t QB = (size_t)NPR * HKP * 2;       // 4,194,304
  _Float16* Xq   = (_Float16*)carve(XB);
  _Float16* Xk   = (_Float16*)carve(XB);
  _Float16* Xv   = (_Float16*)carve(XB);
  _Float16* Wq16 = (_Float16*)carve(WB);
  _Float16* Wk16 = (_Float16*)carve(WB);
  _Float16* Wv16 = (_Float16*)carve(WB);
  _Float16* Wo16 = (_Float16*)carve(WB);
  _Float16* Qp   = (_Float16*)carve(QB);         // [16384,128]
  _Float16* Kp   = (_Float16*)carve(QB);         // [16384,128]
  _Float16* Vt   = (_Float16*)carve(QB);         // [128,16384]
  _Float16* CtxG = (_Float16*)carve(XB);         // [4096,416]

  // allow >64KB dynamic LDS for the fused scores+softmax kernel
  const int smemBytes = 16 * 4096 * 4 + 256 * 4 + 16 * 4 + 16 * 4;
  (void)hipFuncSetAttribute(reinterpret_cast<const void*>(&attn_kernel),
                            hipFuncAttributeMaxDynamicSharedMemorySize, smemBytes);

  // 0) conversions + zero pads (every call: ws is not preserved)
  {
    int nX = NSEQ * DPAD, nW = DMODEL * DPAD;
    int gX = (nX + 255) / 256, gW = (nW + 255) / 256;
    cvt_pad_kernel<<<gX, 256, 0, stream>>>(query, Xq, NSEQ, DMODEL, DPAD);
    cvt_pad_kernel<<<gX, 256, 0, stream>>>(key,   Xk, NSEQ, DMODEL, DPAD);
    cvt_pad_kernel<<<gX, 256, 0, stream>>>(value, Xv, NSEQ, DMODEL, DPAD);
    cvt_pad_kernel<<<gW, 256, 0, stream>>>(Wq, Wq16, DMODEL, DMODEL, DPAD);
    cvt_pad_kernel<<<gW, 256, 0, stream>>>(Wk, Wk16, DMODEL, DMODEL, DPAD);
    cvt_pad_kernel<<<gW, 256, 0, stream>>>(Wv, Wv16, DMODEL, DMODEL, DPAD);
    cvt_pad_kernel<<<gW, 256, 0, stream>>>(Wo, Wo16, DMODEL, DMODEL, DPAD);
    size_t zb = 3 * QB + XB;                     // Qp,Kp,Vt,CtxG
    int n16 = (int)(zb / 16);
    zero16_kernel<<<(n16 + 255) / 256, 256, 0, stream>>>((uint4*)Qp, n16);
  }

  // 1) projections
  proj_kernel<<<256, 128, 0, stream>>>(Xq, Wq16, bq, Qp, nullptr);
  proj_kernel<<<256, 128, 0, stream>>>(Xk, Wk16, bk, Kp, nullptr);
  proj_kernel<<<256, 128, 0, stream>>>(Xv, Wv16, bv, nullptr, Vt);

  // 2) scores + softmax -> attn (fp32 output, second element of return tuple)
  attn_kernel<<<dim3(256, NHEAD), 256, smemBytes, stream>>>(Qp, Kp, attn);

  // 3) ctx = attn @ v  -> CtxG in reshaped-back [4096,416] f16 layout
  ctx_kernel<<<dim3(64, NHEAD), 128, 0, stream>>>(attn, Vt, CtxG);

  // 4) output projection + residual + LayerNorm -> out
  outln_kernel<<<256, 128, 0, stream>>>(CtxG, Wo16, bo, query, gamma, beta, out);
}
